// DBlock_22376779612501
// MI455X (gfx1250) — matrix operand order repaired and checked by reading, not compiled, not run on hardware
//
#include <hip/hip_runtime.h>

// ---------------------------------------------------------------------------
// Problem constants (from reference: N=32, C_IN=16, C_OUT=1, H=W=128, DS=5)
// ---------------------------------------------------------------------------
constexpr int N_  = 32;
constexpr int CI_ = 16;
constexpr int H_  = 128;
constexpr int W_  = 128;
constexpr int DS_ = 5;
constexpr int SH_ = 2 * DS_ - 1;   // 9 shifts per axis for autocorr
constexpr int WP_ = 144;           // 8 + 128 + 8 padded width (zero halos)
constexpr int M_  = CI_ * DS_ * DS_;   // 400 normal-equation rows
constexpr int LD_ = 416;           // padded leading dim for augmented [Q|P]

typedef __attribute__((ext_vector_type(16))) __bf16 v16bf;
typedef __attribute__((ext_vector_type(8)))  __bf16 v8bf;
typedef __attribute__((ext_vector_type(8)))  float  v8f;

// ---------------------------------------------------------------------------
// Fragment helpers (constant-index shuffles -> pure register ops)
// ---------------------------------------------------------------------------
__device__ inline v16bf concat8(v8bf a, v8bf b) {
  return __builtin_shufflevector(a, b, 0, 1, 2, 3, 4, 5, 6, 7,
                                       8, 9, 10, 11, 12, 13, 14, 15);
}

template <int DV>
__device__ inline v16bf shiftwin(v16bf a, v16bf b) {
  return __builtin_shufflevector(a, b,
      8 + DV, 9 + DV, 10 + DV, 11 + DV, 12 + DV, 13 + DV, 14 + DV, 15 + DV,
      16 + DV, 17 + DV, 18 + DV, 19 + DV, 20 + DV, 21 + DV, 22 + DV, 23 + DV);
}

// ---------------------------------------------------------------------------
// f32 -> bf16 (RNE) with zero halos of width 8 on each side of W.
// ---------------------------------------------------------------------------
__global__ __launch_bounds__(256)
void cvt_pad_kernel(const float* __restrict__ x, unsigned short* __restrict__ xbp) {
  size_t idx = (size_t)blockIdx.x * 256 + threadIdx.x;
  size_t total = (size_t)N_ * CI_ * H_ * WP_;
  if (idx >= total) return;
  int col = (int)(idx % WP_);
  size_t rowid = idx / WP_;              // ((n*CI + c)*H + h)
  float val = 0.0f;
  int w = col - 8;
  if (w >= 0 && w < W_) val = x[rowid * W_ + w];
  unsigned int u = __float_as_uint(val);
  unsigned int r = u + 0x7FFFu + ((u >> 16) & 1u);   // round-to-nearest-even
  xbp[idx] = (unsigned short)(r >> 16);
}

__global__ __launch_bounds__(256)
void zero_kernel(float* __restrict__ p, int count) {
  int i = blockIdx.x * 256 + threadIdx.x;
  if (i < count) p[i] = 0.0f;
}

// ---------------------------------------------------------------------------
// autocorr via WMMA:
//   R[n][u][v][i][j] = sum_{h,w} x[n,i,h,w] * x[n,j,h+u-4,w+v-4]
// One workgroup per (n,u); 4 waves split the h range; each wave keeps 9
// v-shift f32 accumulators and issues 9 v_wmma_f32_16x16x32_bf16 per K=32
// block, reusing one A fragment and one 32-elem B window (shuffled per v).
// ---------------------------------------------------------------------------
#define ACC_V(IDX, DV)                                                        \
  {                                                                           \
    v16bf bfrag = shiftwin<DV>(wlo, whi);                                     \
    acc[IDX] = __builtin_amdgcn_wmma_f32_16x16x32_bf16(                       \
        false, afrag, false, bfrag, (short)0, acc[IDX], false, false);        \
  }

__global__ __launch_bounds__(128)
void autocorr_wmma_kernel(const unsigned short* __restrict__ xbp_raw,
                          float* __restrict__ R) {
  const __bf16* xbp = (const __bf16*)xbp_raw;
  int wg   = blockIdx.x;           // N_*SH_ workgroups
  int n    = wg / SH_;
  int u    = wg % SH_;
  int du   = u - (DS_ - 1);        // -4..4 row shift (wave-uniform)
  int wave = threadIdx.x >> 5;
  int lane = threadIdx.x & 31;
  int half = lane >> 4;
  int lrow = lane & 15;            // A row i / B column j held by this lane

  const __bf16* xb_n = xbp + (size_t)n * CI_ * H_ * WP_;

  v8f acc[SH_];
#pragma unroll
  for (int v = 0; v < SH_; ++v)
#pragma unroll
    for (int e = 0; e < 8; ++e) acc[v][e] = 0.0f;

  for (int h = wave * 32; h < wave * 32 + 32; ++h) {
    int hs = h + du;
    if (hs < 0 || hs >= H_) continue;           // zero-padded rows: skip
    const __bf16* arow = xb_n + ((size_t)lrow * H_ + h)  * WP_;
    const __bf16* brow = xb_n + ((size_t)lrow * H_ + hs) * WP_;
#pragma unroll
    for (int kb = 0; kb < 4; ++kb) {
      int k0 = kb * 32;
      // A fragment: row i = lrow, K-chunks {0..15} / {16..31}
      v8bf c0 = *(const v8bf*)(arow + 8  + k0 + half * 16);
      v8bf c1 = *(const v8bf*)(arow + 16 + k0 + half * 16);
      v16bf afrag = concat8(c0, c1);
      // B window: 32 contiguous bf16 covering all 9 dv shifts (aligned 32B)
      const v16bf* q = (const v16bf*)(brow + k0 + 16 * half);
      v16bf wlo = q[0];
      v16bf whi = q[1];
      ACC_V(0, -4) ACC_V(1, -3) ACC_V(2, -2) ACC_V(3, -1) ACC_V(4, 0)
      ACC_V(5,  1) ACC_V(6,  2) ACC_V(7,  3) ACC_V(8,  4)
    }
  }

  // C layout: VGPR r -> row (r + 8*half), column lrow. Merge 4 wave slices.
  float* Rn = R + (((size_t)n * SH_ + u) * SH_) * 256;
#pragma unroll
  for (int v = 0; v < SH_; ++v) {
    float* Rv = Rn + (size_t)v * 256;
#pragma unroll
    for (int r = 0; r < 8; ++r) {
      int i = r + 8 * half;
      __hip_atomic_fetch_add(&Rv[i * 16 + lrow], acc[v][r],
                             __ATOMIC_RELAXED, __HIP_MEMORY_SCOPE_AGENT);
    }
  }
}

// ---------------------------------------------------------------------------
// crosscorr: P[n][j*25+u*5+v] = sum_{h,w} y[n,0,h,w]*x[n,j,h+u-2,w+v-2]
// ---------------------------------------------------------------------------
__global__ __launch_bounds__(256)
void crosscorr_kernel(const float* __restrict__ x, const float* __restrict__ y,
                      float* __restrict__ Pb) {
  int n = blockIdx.x >> 4;
  int j = blockIdx.x & 15;
  const float* yp = y + (size_t)n * H_ * W_;
  const float* xp = x + ((size_t)n * CI_ + j) * H_ * W_;
  float av[25];
#pragma unroll
  for (int s = 0; s < 25; ++s) av[s] = 0.0f;

  for (int p = threadIdx.x; p < H_ * W_; p += 256) {
    int h = p >> 7, w = p & 127;
    float yv = yp[p];
#pragma unroll
    for (int u = 0; u < DS_; ++u) {
      int hh = h + u - 2;
      if (hh < 0 || hh >= H_) continue;
#pragma unroll
      for (int v = 0; v < DS_; ++v) {
        int ww = w + v - 2;
        if (ww < 0 || ww >= W_) continue;
        av[u * DS_ + v] += yv * xp[hh * W_ + ww];
      }
    }
  }
  __shared__ float red[256];
  for (int s = 0; s < 25; ++s) {
    red[threadIdx.x] = av[s];
    __syncthreads();
    for (int off = 128; off > 0; off >>= 1) {
      if (threadIdx.x < (unsigned)off) red[threadIdx.x] += red[threadIdx.x + off];
      __syncthreads();
    }
    if (threadIdx.x == 0) Pb[(size_t)n * M_ + j * 25 + s] = red[0];
    __syncthreads();
  }
}

// ---------------------------------------------------------------------------
// Block-Toeplitz assembly of augmented [Q | P], with ridge a*I and a*d on P.
//   Qa[n][(i,p,q)][(j,oi,oj)] = R[n][4+oi-p][4+oj-q][i][j] (+ a on diag)
//   Qa[n][row][400]           = P[n][row] + a * d[n][row]
// ---------------------------------------------------------------------------
__global__ __launch_bounds__(256)
void assemble_kernel(const float* __restrict__ R, const float* __restrict__ Pb,
                     const float* __restrict__ dcur, const float* __restrict__ alpha,
                     const float* __restrict__ reg, float* __restrict__ Qa) {
  size_t idx = (size_t)blockIdx.x * 256 + threadIdx.x;
  size_t total = (size_t)N_ * M_ * (M_ + 1);
  if (idx >= total) return;
  int n   = (int)(idx / ((size_t)M_ * (M_ + 1)));
  int rem = (int)(idx % ((size_t)M_ * (M_ + 1)));
  int row = rem / (M_ + 1);
  int col = rem % (M_ + 1);
  float a = alpha[n] * (float)(H_ * W_) * reg[0] / (float)(DS_ * DS_ * CI_);
  float val;
  if (col < M_) {
    int i  = row / 25, pq = row % 25, p  = pq / 5, q  = pq % 5;
    int jj = col / 25, oo = col % 25, oi = oo / 5, oj = oo % 5;
    int u = (DS_ - 1) + oi - p;
    int v = (DS_ - 1) + oj - q;
    val = R[((((size_t)n * SH_ + u) * SH_ + v) * 16 + i) * 16 + jj];
    if (row == col) val += a;
  } else {
    val = Pb[(size_t)n * M_ + row] + a * dcur[(size_t)n * M_ + row];
  }
  Qa[((size_t)n * M_ + row) * LD_ + col] = val;
}

// ---------------------------------------------------------------------------
// Batched solve: one workgroup per n. Gaussian elimination (SPD+ridge: no
// pivoting needed), pivot row cached in LDS; then LDS-reduced back-subst.
// ---------------------------------------------------------------------------
__global__ __launch_bounds__(256)
void solve_kernel(float* __restrict__ Qa, float* __restrict__ D) {
  int n = blockIdx.x;
  float* Mx = Qa + (size_t)n * M_ * LD_;
  __shared__ float prow[M_ + 1];
  __shared__ float xs[M_];
  __shared__ float red[256];
  int tid = threadIdx.x;

  for (int k = 0; k < M_ - 1; ++k) {
    for (int c = k + tid; c <= M_; c += 256) prow[c] = Mx[(size_t)k * LD_ + c];
    __syncthreads();
    float inv = 1.0f / prow[k];
    for (int r = k + 1 + tid; r < M_; r += 256) {
      float f = Mx[(size_t)r * LD_ + k] * inv;
      float* Mr = Mx + (size_t)r * LD_;
      for (int c = k + 1; c <= M_; ++c) Mr[c] -= f * prow[c];
    }
    __syncthreads();
  }
  for (int r = M_ - 1; r >= 0; --r) {
    float s = 0.0f;
    for (int c = r + 1 + tid; c < M_; c += 256) s += Mx[(size_t)r * LD_ + c] * xs[c];
    red[tid] = s;
    __syncthreads();
    for (int off = 128; off > 0; off >>= 1) {
      if (tid < off) red[tid] += red[tid + off];
      __syncthreads();
    }
    if (tid == 0)
      xs[r] = (Mx[(size_t)r * LD_ + M_] - red[0]) / Mx[(size_t)r * LD_ + r];
    __syncthreads();
  }
  for (int r = tid; r < M_; r += 256) D[(size_t)n * M_ + r] = xs[r];
}

// ---------------------------------------------------------------------------
// h0[n][17][25]: channels 0..15 = solved D2, channel 16 = 1/sqrt(beta[n])
// ---------------------------------------------------------------------------
__global__ __launch_bounds__(256)
void build_h0_kernel(const float* __restrict__ D2, const float* __restrict__ beta,
                     float* __restrict__ h0) {
  int idx = blockIdx.x * 256 + threadIdx.x;
  int total = N_ * 17 * 25;
  if (idx >= total) return;
  int n = idx / (17 * 25);
  int rem = idx % (17 * 25);
  int ch = rem / 25, s = rem % 25;
  h0[idx] = (ch < 16) ? D2[(size_t)n * M_ + ch * 25 + s] : rsqrtf(beta[n]);
}

// ---------------------------------------------------------------------------
// 3x3 conv, pad 1, on 5x5 maps; inputs + weights staged in LDS; optional
// ReLU and optional residual (h0[:, :16]) add for the final layer.
// ---------------------------------------------------------------------------
__global__ __launch_bounds__(256)
void conv3x3_kernel(const float* __restrict__ in, const float* __restrict__ w,
                    const float* __restrict__ b, const float* __restrict__ resid,
                    float* __restrict__ out, int cin, int relu) {
  int n = blockIdx.x;
  __shared__ float sin_[17 * 25];
  __shared__ float sw_[16 * 17 * 9];
  int tid = threadIdx.x;
  int insz = cin * 25;
  for (int i = tid; i < insz; i += 256) sin_[i] = in[(size_t)n * insz + i];
  int wsz = 16 * cin * 9;
  for (int i = tid; i < wsz; i += 256) sw_[i] = w[i];
  __syncthreads();
  for (int o = tid; o < 16 * 25; o += 256) {
    int co = o / 25, s = o % 25, hh = s / 5, ww = s % 5;
    float acc = b[co];
    for (int ci = 0; ci < cin; ++ci) {
      const float* si = sin_ + ci * 25;
      const float* wk = sw_ + (co * cin + ci) * 9;
#pragma unroll
      for (int kh = 0; kh < 3; ++kh) {
        int ih = hh + kh - 1;
        if (ih < 0 || ih >= 5) continue;
#pragma unroll
        for (int kw = 0; kw < 3; ++kw) {
          int iw = ww + kw - 1;
          if (iw < 0 || iw >= 5) continue;
          acc += si[ih * 5 + iw] * wk[kh * 3 + kw];
        }
      }
    }
    if (relu) acc = acc > 0.0f ? acc : 0.0f;
    if (resid) acc += resid[(size_t)n * (17 * 25) + co * 25 + s];
    out[(size_t)n * 400 + o] = acc;
  }
}

// ---------------------------------------------------------------------------
// Launch: two DSB passes (cvt -> zero -> WMMA autocorr -> crosscorr ->
// assemble -> solve), then h0 + 6 convs.
// ---------------------------------------------------------------------------
extern "C" void kernel_launch(void* const* d_in, const int* in_sizes, int n_in,
                              void* d_out, int out_size, void* d_ws, size_t ws_size,
                              hipStream_t stream) {
  (void)in_sizes; (void)n_in; (void)out_size; (void)ws_size;
  const float* x1    = (const float*)d_in[0];
  const float* x2    = (const float*)d_in[1];
  const float* d0    = (const float*)d_in[2];
  const float* y1    = (const float*)d_in[3];
  const float* y2    = (const float*)d_in[4];
  const float* alpha = (const float*)d_in[5];
  const float* beta  = (const float*)d_in[6];
  const float* reg   = (const float*)d_in[7];
  const float* w[6];
  const float* b[6];
  for (int i = 0; i < 6; ++i) {
    w[i] = (const float*)d_in[8 + 2 * i];
    b[i] = (const float*)d_in[9 + 2 * i];
  }

  char* ws = (char*)d_ws;
  size_t off = 0;
  auto take = [&](size_t bytes) -> void* {
    void* p = ws + off;
    off = (off + bytes + 255) & ~(size_t)255;
    return p;
  };
  // xbp (bf16 padded x, 18.9 MB) and Qa (augmented system, 21.3 MB) have
  // disjoint lifetimes within a pass -> alias them in one region.
  size_t big = (size_t)N_ * M_ * LD_ * sizeof(float);   // >= xbp bytes
  void* shared_region = take(big);
  unsigned short* xbp = (unsigned short*)shared_region;
  float* Qa = (float*)shared_region;
  float* R  = (float*)take((size_t)N_ * SH_ * SH_ * 256 * sizeof(float));
  float* Pb = (float*)take((size_t)N_ * M_ * sizeof(float));
  float* D1 = (float*)take((size_t)N_ * M_ * sizeof(float));
  float* D2 = (float*)take((size_t)N_ * M_ * sizeof(float));
  float* h0 = (float*)take((size_t)N_ * 17 * 25 * sizeof(float));
  float* t1 = (float*)take((size_t)N_ * 16 * 25 * sizeof(float));
  float* t2 = (float*)take((size_t)N_ * 16 * 25 * sizeof(float));

  auto dsb_pass = [&](const float* x, const float* y, const float* dprev,
                      float* Dout) {
    int cvt_total = N_ * CI_ * H_ * WP_;
    cvt_pad_kernel<<<(cvt_total + 255) / 256, 256, 0, stream>>>(x, xbp);
    int rcount = N_ * SH_ * SH_ * 256;
    zero_kernel<<<(rcount + 255) / 256, 256, 0, stream>>>(R, rcount);
    autocorr_wmma_kernel<<<N_ * SH_, 128, 0, stream>>>(xbp, R);
    crosscorr_kernel<<<N_ * CI_, 256, 0, stream>>>(x, y, Pb);
    int atotal = N_ * M_ * (M_ + 1);
    assemble_kernel<<<(atotal + 255) / 256, 256, 0, stream>>>(R, Pb, dprev,
                                                              alpha, reg, Qa);
    solve_kernel<<<N_, 256, 0, stream>>>(Qa, Dout);
  };
  dsb_pass(x1, y1, d0, D1);
  dsb_pass(x2, y2, D1, D2);

  int htotal = N_ * 17 * 25;
  build_h0_kernel<<<(htotal + 255) / 256, 256, 0, stream>>>(D2, beta, h0);
  conv3x3_kernel<<<N_, 256, 0, stream>>>(h0, w[0], b[0], nullptr, t1, 17, 1);
  conv3x3_kernel<<<N_, 256, 0, stream>>>(t1, w[1], b[1], nullptr, t2, 16, 1);
  conv3x3_kernel<<<N_, 256, 0, stream>>>(t2, w[2], b[2], nullptr, t1, 16, 1);
  conv3x3_kernel<<<N_, 256, 0, stream>>>(t1, w[3], b[3], nullptr, t2, 16, 1);
  conv3x3_kernel<<<N_, 256, 0, stream>>>(t2, w[4], b[4], nullptr, t1, 16, 1);
  conv3x3_kernel<<<N_, 256, 0, stream>>>(t1, w[5], b[5], h0, (float*)d_out, 16, 0);
}
